// LSTMDSTGCRN_83846351552832
// MI455X (gfx1250) — matrix operand order repaired and checked by reading, not compiled, not run on hardware
//
#include <hip/hip_runtime.h>

// ---------------- problem constants ----------------
#define BB 32
#define TT 24
#define NN 2048
#define CIN 2
#define HID 64
#define EMB 16
#define HORIZON 12

#define C1 (CIN + HID)      // 66 (true K of cell-1 gates)
#define C1P 96              // padded to multiple of 32 for WMMA
#define C2 (HID + HID)      // 128
#define NC1P (BB * C1P)     // 3072
#define NC2 (BB * C2)       // 4096
#define MROWS (NN * BB)     // 65536

// ---------------- types ----------------
typedef __attribute__((ext_vector_type(16))) __bf16 v16bf;
typedef __attribute__((ext_vector_type(8)))  float  v8f;
typedef __attribute__((ext_vector_type(4)))  unsigned int uivec4;
typedef __attribute__((ext_vector_type(4)))  unsigned int u32x4v;
typedef __attribute__((ext_vector_type(8)))  int          i32x8v;
typedef __attribute__((ext_vector_type(4)))  int          i32x4v;

__device__ __forceinline__ unsigned short f2bf(float f) {
    unsigned int u = __builtin_bit_cast(unsigned int, f);
    unsigned int r = u + 0x7FFFu + ((u >> 16) & 1u);   // round-to-nearest-even
    return (unsigned short)(r >> 16);
}
__device__ __forceinline__ float bf2f(unsigned short s) {
    unsigned int u = ((unsigned int)s) << 16;
    return __builtin_bit_cast(float, u);
}

// ---------------- TDM helpers ----------------
// Low 32 bits of a generic LDS pointer ARE the LDS byte offset (aperture rule).
__device__ __forceinline__ unsigned lds_off(const void* p) {
    return (unsigned)(unsigned long long)p;
}

__device__ __forceinline__ void tdm_issue(u32x4v g0, i32x8v g1) {
    i32x4v z4 = {0, 0, 0, 0};
#if defined(__clang_major__) && (__clang_major__ >= 23)
    i32x8v z8 = {0, 0, 0, 0, 0, 0, 0, 0};
    __builtin_amdgcn_tensor_load_to_lds(g0, g1, z4, z4, z8, 0);
#else
    __builtin_amdgcn_tensor_load_to_lds(g0, g1, z4, z4, 0);
#endif
}

// 2D bf16 tile load: tile_dim0 = 32 elements (64B = 16 DWORDs) along the row,
// tile_dim1 = tile_rows, row stride = row_len elements. LDS padding: after each
// 16 DWORDs insert 4 DWORDs -> LDS row stride 40 shorts (bank-conflict-free).
__device__ __forceinline__ void tdm_load_tile32(const unsigned short* tile_ptr,
                                                unsigned lds_addr,
                                                int tile_rows, int row_len, int tot_rows) {
    unsigned long long ga = (unsigned long long)tile_ptr;
    u32x4v g0 = { 1u,                                    // count=1
                  lds_addr,
                  (unsigned)ga,
                  (unsigned)((ga >> 32) & 0x01FFFFFFu) | 0x80000000u };  // type=2
    i32x8v g1;
    g1[0] = (1 << 16)       // data_size = 2 bytes
          | (1 << 20)       // pad_enable
          | (3 << 22)       // pad_interval = 16 DWORDs
          | (3 << 25);      // pad_amount   = 4 DWORDs
    g1[1] = (int)(((unsigned)row_len & 0xFFFFu) << 16);                 // tensor_dim0 lo
    g1[2] = (int)(((unsigned)row_len >> 16) |
                  (((unsigned)tot_rows & 0xFFFFu) << 16));              // dim0 hi | dim1 lo
    g1[3] = (int)(((unsigned)tot_rows >> 16) | (32u << 16));            // dim1 hi | tile_dim0=32
    g1[4] = (int)((unsigned)tile_rows & 0xFFFFu);                       // tile_dim1
    g1[5] = row_len;                                                    // dim0_stride lo32
    g1[6] = 0;
    g1[7] = 0;
    tdm_issue(g0, g1);
}

// ---------------- adjacency: A = softmax(relu(E E^T)) row-wise, bf16 out ----
__global__ __launch_bounds__(256)
void adj_kernel(const float* __restrict__ E, unsigned short* __restrict__ A) {
    __shared__ float er[EMB];
    __shared__ float red[256];
    const int n = blockIdx.x;
    const int tid = threadIdx.x;
    if (tid < EMB) er[tid] = E[n * EMB + tid];
    __syncthreads();

    float sc[8];
    float lmax = 0.0f;
    #pragma unroll
    for (int j = 0; j < 8; ++j) {
        const int m = tid + j * 256;
        const float* em = E + m * EMB;
        float d = 0.0f;
        #pragma unroll
        for (int i = 0; i < EMB; ++i) d += er[i] * em[i];
        d = fmaxf(d, 0.0f);
        sc[j] = d;
        lmax = fmaxf(lmax, d);
    }
    red[tid] = lmax; __syncthreads();
    for (int s = 128; s > 0; s >>= 1) {
        if (tid < s) red[tid] = fmaxf(red[tid], red[tid + s]);
        __syncthreads();
    }
    const float bmax = red[0];
    __syncthreads();

    float lsum = 0.0f;
    #pragma unroll
    for (int j = 0; j < 8; ++j) { sc[j] = __expf(sc[j] - bmax); lsum += sc[j]; }
    red[tid] = lsum; __syncthreads();
    for (int s = 128; s > 0; s >>= 1) {
        if (tid < s) red[tid] += red[tid + s];
        __syncthreads();
    }
    const float inv = 1.0f / red[0];
    #pragma unroll
    for (int j = 0; j < 8; ++j)
        A[(size_t)n * NN + tid + j * 256] = f2bf(sc[j] * inv);
}

// ---------------- pack x_t into combT1 (bf16, transposed [B*C1P x N]) -------
__global__ void pack_x(const float* __restrict__ x, unsigned short* __restrict__ combT1, int t) {
    const int idx = blockIdx.x * blockDim.x + threadIdx.x;
    if (idx >= MROWS) return;
    const int b = idx >> 11;
    const int n = idx & (NN - 1);
    const float* xp = x + (((size_t)b * TT + t) * NN + n) * CIN;
    combT1[((size_t)(b * C1P + 0)) * NN + n] = f2bf(xp[0]);
    combT1[((size_t)(b * C1P + 1)) * NN + n] = f2bf(xp[1]);
}

// ---------------- W transpose+pad into bf16 [256 x Kp] ----------------
__global__ void wprep_kernel(const float* __restrict__ W, unsigned short* __restrict__ WT,
                             int C, int Kp) {
    const int idx = blockIdx.x * blockDim.x + threadIdx.x;
    if (idx >= 256 * Kp) return;
    const int col = idx / Kp;
    const int k = idx % Kp;
    WT[(size_t)col * Kp + k] = (k < C) ? f2bf(W[(size_t)k * 256 + col]) : (unsigned short)0;
}

// ---------------- WMMA GEMM with TDM tile feeds ----------------
// C[M,Nc] = A[M,K] * B^T[Nc,K]   (A row-major bf16, B given transposed bf16)
// Block tile 64(M) x 128(N); 8 waves, each 32x32 (4 WMMA accumulators).
// Both LDS tiles loaded by the Tensor Data Mover (wave 0 issues descriptors,
// TENSORcnt + workgroup barrier publishes each stage). K chunk = 32.
template<bool OUT_BF16>
__global__ __launch_bounds__(256)
void gemm_tdm_wmma(const unsigned short* __restrict__ Abuf,  // [M x K]
                   const unsigned short* __restrict__ Bbuf,  // [Nc x K]
                   void* __restrict__ Cout,                  // [M x Nc]
                   int M, int Nc, int K) {
    __shared__ __align__(16) unsigned short As[2][64][40];   // [row][k]
    __shared__ __align__(16) unsigned short Bs[2][128][40];  // [col][k]

    const int tid = threadIdx.x;
    const int wave = tid >> 5;
    const int lane = tid & 31;
    const int bm = blockIdx.y * 64;
    const int bn = blockIdx.x * 128;
    const int wm = (wave & 1) * 32;
    const int wn = (wave >> 1) * 32;
    const int half = lane >> 4;
    const int l16 = lane & 31 & 15;
    const int KT = K / 32;

    v8f a00 = {0.f,0.f,0.f,0.f,0.f,0.f,0.f,0.f};
    v8f a01 = a00, a10 = a00, a11 = a00;

    const unsigned short* Atile = Abuf + (size_t)bm * K;
    const unsigned short* Btile = Bbuf + (size_t)bn * K;

    if (wave == 0) {
        tdm_load_tile32(Atile, lds_off(&As[0][0][0]), 64, K, M);
        tdm_load_tile32(Btile, lds_off(&Bs[0][0][0]), 128, K, Nc);
        if (KT > 1) {
            tdm_load_tile32(Atile + 32, lds_off(&As[1][0][0]), 64, K, M);
            tdm_load_tile32(Btile + 32, lds_off(&Bs[1][0][0]), 128, K, Nc);
            __builtin_amdgcn_s_wait_tensorcnt(2);   // stage 0 complete (in-order)
        } else {
            __builtin_amdgcn_s_wait_tensorcnt(0);
        }
    }
    __syncthreads();

    for (int kt = 0; kt < KT; ++kt) {
        const int cur = kt & 1;

        union { uivec4 u[2]; v16bf v; } fa0, fa1, fb0, fb1;
        const unsigned short* ap0 = &As[cur][wm + l16][0];
        fa0.u[0] = *(const uivec4*)(ap0 + half * 8);
        fa0.u[1] = *(const uivec4*)(ap0 + 16 + half * 8);
        const unsigned short* ap1 = &As[cur][wm + 16 + l16][0];
        fa1.u[0] = *(const uivec4*)(ap1 + half * 8);
        fa1.u[1] = *(const uivec4*)(ap1 + 16 + half * 8);
        const unsigned short* bp0 = &Bs[cur][wn + l16][half * 8];
        fb0.u[0] = *(const uivec4*)(bp0);
        fb0.u[1] = *(const uivec4*)(bp0 + 16);
        const unsigned short* bp1 = &Bs[cur][wn + 16 + l16][half * 8];
        fb1.u[0] = *(const uivec4*)(bp1);
        fb1.u[1] = *(const uivec4*)(bp1 + 16);

        a00 = __builtin_amdgcn_wmma_f32_16x16x32_bf16(false, fa0.v, false, fb0.v, (short)0, a00, false, false);
        a01 = __builtin_amdgcn_wmma_f32_16x16x32_bf16(false, fa0.v, false, fb1.v, (short)0, a01, false, false);
        a10 = __builtin_amdgcn_wmma_f32_16x16x32_bf16(false, fa1.v, false, fb0.v, (short)0, a10, false, false);
        a11 = __builtin_amdgcn_wmma_f32_16x16x32_bf16(false, fa1.v, false, fb1.v, (short)0, a11, false, false);

        __syncthreads();                       // everyone done reading buffer `cur`
        if (wave == 0) {
            if (kt + 2 < KT) {                 // prefetch 2 ahead into `cur`
                const int k0 = (kt + 2) * 32;
                tdm_load_tile32(Atile + k0, lds_off(&As[cur][0][0]), 64, K, M);
                tdm_load_tile32(Btile + k0, lds_off(&Bs[cur][0][0]), 128, K, Nc);
                __builtin_amdgcn_s_wait_tensorcnt(2);   // stage kt+1 complete
            } else {
                __builtin_amdgcn_s_wait_tensorcnt(0);
            }
        }
        __syncthreads();                       // next buffer published
    }

    // D layout: vgpr r, lane l -> row = r + 8*(l>>4), col = l&15
    #pragma unroll
    for (int r = 0; r < 8; ++r) {
        const int row0 = bm + wm + r + half * 8;
        const int col0 = bn + wn + l16;
        if (OUT_BF16) {
            unsigned short* C = (unsigned short*)Cout;
            C[(size_t)row0 * Nc + col0]             = f2bf(a00[r]);
            C[(size_t)row0 * Nc + col0 + 16]        = f2bf(a01[r]);
            C[(size_t)(row0 + 16) * Nc + col0]      = f2bf(a10[r]);
            C[(size_t)(row0 + 16) * Nc + col0 + 16] = f2bf(a11[r]);
        } else {
            float* C = (float*)Cout;
            C[(size_t)row0 * Nc + col0]             = a00[r];
            C[(size_t)row0 * Nc + col0 + 16]        = a01[r];
            C[(size_t)(row0 + 16) * Nc + col0]      = a10[r];
            C[(size_t)(row0 + 16) * Nc + col0 + 16] = a11[r];
        }
    }
}

// ---------------- pointwise LSTM: gates -> (c,h), h transposed via LDS ------
// gates: [(n*B+b) x 256] f32 (i|f|g|o). Block handles fixed b, 64 k x 64 n.
__global__ __launch_bounds__(256)
void pointwise_kernel(const float* __restrict__ gates, const float* __restrict__ bias,
                      float* __restrict__ cst,
                      unsigned short* __restrict__ d0, int d0C, int d0off,
                      unsigned short* __restrict__ d1, int d1C, int d1off) {
    __shared__ __align__(16) unsigned short ht[64][72];   // [k][n], padded
    const int tid = threadIdx.x;
    const int n0 = blockIdx.x * 64;
    const int b = blockIdx.y;

    #pragma unroll
    for (int j = 0; j < 16; ++j) {
        const int flat = tid + j * 256;          // 0..4095
        const int k = flat & 63;
        const int nl = flat >> 6;
        const size_t r = (size_t)(n0 + nl) * BB + b;
        const float* g = gates + r * 256;
        const float ai = g[k]       + bias[k];
        const float af = g[64 + k]  + bias[64 + k];
        const float ag = g[128 + k] + bias[128 + k];
        const float ao = g[192 + k] + bias[192 + k];
        const float si = 1.0f / (1.0f + __expf(-ai));
        const float sf = 1.0f / (1.0f + __expf(-af));
        const float so = 1.0f / (1.0f + __expf(-ao));
        const float cn = sf * cst[r * 64 + k] + si * tanhf(ag);
        cst[r * 64 + k] = cn;
        ht[k][nl] = f2bf(so * tanhf(cn));
    }
    __syncthreads();

    const int kr = tid >> 2;          // 0..63
    const int seg = (tid & 3) * 16;   // 0,16,32,48
    uivec4 v0 = *(const uivec4*)&ht[kr][seg];
    uivec4 v1 = *(const uivec4*)&ht[kr][seg + 8];
    {
        unsigned short* p = d0 + ((size_t)(b * d0C + d0off + kr)) * NN + n0 + seg;
        *(uivec4*)p = v0;
        *(uivec4*)(p + 8) = v1;
    }
    if (d1) {
        unsigned short* p = d1 + ((size_t)(b * d1C + d1off + kr)) * NN + n0 + seg;
        *(uivec4*)p = v0;
        *(uivec4*)(p + 8) = v1;
    }
}

// ---------------- final projection: out = h2 @ Wp + bp ----------------
__global__ void proj_kernel(const unsigned short* __restrict__ combT2,
                            const float* __restrict__ Wp,  // [64 x 12]
                            const float* __restrict__ bp,  // [12]
                            float* __restrict__ out) {     // [B,12,N,1]
    const int idx = blockIdx.x * blockDim.x + threadIdx.x;
    if (idx >= MROWS) return;
    const int b = idx >> 11;
    const int n = idx & (NN - 1);
    float acc[HORIZON];
    #pragma unroll
    for (int h = 0; h < HORIZON; ++h) acc[h] = bp[h];
    for (int k = 0; k < HID; ++k) {
        const float hv = bf2f(combT2[((size_t)(b * C2 + HID + k)) * NN + n]);
        #pragma unroll
        for (int h = 0; h < HORIZON; ++h) acc[h] += hv * Wp[k * HORIZON + h];
    }
    #pragma unroll
    for (int h = 0; h < HORIZON; ++h)
        out[((size_t)(b * HORIZON + h)) * NN + n] = acc[h];
}

// ---------------- workspace layout ----------------
#define ALIGN256(x) (((x) + 255) & ~(size_t)255)
static const size_t SZ_A      = (size_t)NN * NN * 2;       // bf16 adjacency
static const size_t SZ_COMBT1 = (size_t)NC1P * NN * 2;     // bf16, transposed
static const size_t SZ_COMBT2 = (size_t)NC2 * NN * 2;      // bf16, transposed
static const size_t SZ_CONV   = (size_t)NN * NC2 * 2;      // bf16 (max of both cells)
static const size_t SZ_GATES  = (size_t)MROWS * 256 * 4;   // f32
static const size_t SZ_C      = (size_t)MROWS * HID * 4;   // f32 cell state
static const size_t SZ_W1T    = (size_t)256 * C1P * 2;
static const size_t SZ_W2T    = (size_t)256 * C2 * 2;

extern "C" void kernel_launch(void* const* d_in, const int* in_sizes, int n_in,
                              void* d_out, int out_size, void* d_ws, size_t ws_size,
                              hipStream_t stream) {
    const float* x  = (const float*)d_in[0];
    const float* E  = (const float*)d_in[1];
    const float* W1 = (const float*)d_in[2];
    const float* b1 = (const float*)d_in[3];
    const float* W2 = (const float*)d_in[4];
    const float* b2 = (const float*)d_in[5];
    const float* Wp = (const float*)d_in[6];
    const float* bp = (const float*)d_in[7];
    float* out = (float*)d_out;

    char* ws = (char*)d_ws;
    size_t off = 0;
    unsigned short* A      = (unsigned short*)(ws + off); off += ALIGN256(SZ_A);
    unsigned short* combT1 = (unsigned short*)(ws + off); off += ALIGN256(SZ_COMBT1);
    unsigned short* combT2 = (unsigned short*)(ws + off); off += ALIGN256(SZ_COMBT2);
    unsigned short* convB  = (unsigned short*)(ws + off); off += ALIGN256(SZ_CONV);
    float* gates = (float*)(ws + off); off += ALIGN256(SZ_GATES);
    float* c1    = (float*)(ws + off); off += ALIGN256(SZ_C);
    float* c2    = (float*)(ws + off); off += ALIGN256(SZ_C);
    unsigned short* W1T = (unsigned short*)(ws + off); off += ALIGN256(SZ_W1T);
    unsigned short* W2T = (unsigned short*)(ws + off); off += ALIGN256(SZ_W2T);

    // zero recurrent state every call (deterministic; graph-capture safe).
    // combT1 zeroing also establishes the K-padding columns (66..95) as zeros.
    hipMemsetAsync(combT1, 0, SZ_COMBT1, stream);
    hipMemsetAsync(combT2, 0, SZ_COMBT2, stream);
    hipMemsetAsync(c1, 0, SZ_C, stream);
    hipMemsetAsync(c2, 0, SZ_C, stream);

    adj_kernel<<<NN, 256, 0, stream>>>(E, A);
    wprep_kernel<<<(256 * C1P + 255) / 256, 256, 0, stream>>>(W1, W1T, C1, C1P);
    wprep_kernel<<<(256 * C2 + 255) / 256, 256, 0, stream>>>(W2, W2T, C2, C2);

    for (int t = 0; t < TT; ++t) {
        pack_x<<<(MROWS + 255) / 256, 256, 0, stream>>>(x, combT1, t);

        // cell 1: conv = A @ comb1   ([2048x2048] x [2048x3072], bf16 out)
        gemm_tdm_wmma<true><<<dim3(NC1P / 128, NN / 64), 256, 0, stream>>>(
            A, combT1, convB, NN, NC1P, NN);
        // gates = conv @ W1          ([65536x96] x [96x256], f32 out)
        gemm_tdm_wmma<false><<<dim3(256 / 128, MROWS / 64), 256, 0, stream>>>(
            convB, W1T, gates, MROWS, 256, C1P);
        // pointwise LSTM; h1 -> combT1 (c=2..65) and combT2 (c=0..63)
        pointwise_kernel<<<dim3(NN / 64, BB), 256, 0, stream>>>(
            gates, b1, c1, combT1, C1P, CIN, combT2, C2, 0);

        // cell 2: conv = A @ comb2   ([2048x2048] x [2048x4096], bf16 out)
        gemm_tdm_wmma<true><<<dim3(NC2 / 128, NN / 64), 256, 0, stream>>>(
            A, combT2, convB, NN, NC2, NN);
        // gates = conv @ W2          ([65536x128] x [128x256], f32 out)
        gemm_tdm_wmma<false><<<dim3(256 / 128, MROWS / 64), 256, 0, stream>>>(
            convB, W2T, gates, MROWS, 256, C2);
        // pointwise LSTM; h2 -> combT2 (c=64..127)
        pointwise_kernel<<<dim3(NN / 64, BB), 256, 0, stream>>>(
            gates, b2, c2, combT2, C2, HID, (unsigned short*)nullptr, 0, 0);
    }

    proj_kernel<<<(MROWS + 255) / 256, 256, 0, stream>>>(combT2, Wp, bp, out);
}
